// Expert_29996051595645
// MI455X (gfx1250) — matrix operand order repaired and checked by reading, not compiled, not run on hardware
//
#include <hip/hip_runtime.h>

#define EXPERTS 8
#define BM 128    // output tile (WMMA M)
#define BN 128    // batch tile (WMMA N)
#define BK 64     // K' elements staged per step (2 WMMA k-steps)
#define ROWE 72   // padded LDS row stride in elements (144B = 9*16B, conflict-free)
#define TILE_E (BM * ROWE)      // 9216 elements per tile
#define TILE_B (TILE_E * 2)     // 18432 bytes per tile
#define LDS_BYTES (4 * TILE_B)  // 73728: W ping/pong + X ping/pong

#define HAVE_TDM __has_builtin(__builtin_amdgcn_tensor_load_to_lds)

typedef __attribute__((ext_vector_type(16))) __bf16 v16bf;
typedef __attribute__((ext_vector_type(8))) float v8f;
typedef __attribute__((ext_vector_type(4))) float v4f;
typedef __attribute__((ext_vector_type(8))) unsigned int v8u;
typedef __attribute__((ext_vector_type(8))) unsigned short v8us;
typedef __attribute__((ext_vector_type(4))) unsigned int u32x4;
typedef __attribute__((ext_vector_type(8))) int i32x8;
typedef __attribute__((ext_vector_type(4))) int i32x4;

__device__ __forceinline__ unsigned short f2bf(float f) {
  unsigned u = __builtin_bit_cast(unsigned, f);
  u += 0x7FFFu + ((u >> 16) & 1u); // RNE
  return (unsigned short)(u >> 16);
}
__device__ __forceinline__ float bf2f(unsigned short h) {
  return __builtin_bit_cast(float, (unsigned)h << 16);
}
__device__ __forceinline__ v16bf make_frag(uint4 a, uint4 b) {
  v8u u;
  u[0] = a.x; u[1] = a.y; u[2] = a.z; u[3] = a.w;
  u[4] = b.x; u[5] = b.y; u[6] = b.z; u[7] = b.w;
  return __builtin_bit_cast(v16bf, u);
}

// ---------- pre-pass 1: weights f32 -> bf16 ----------
__global__ __launch_bounds__(256)
void cvt_f32_bf16(const float* __restrict__ in, unsigned short* __restrict__ out, int n8) {
  int g = blockIdx.x * 256 + threadIdx.x;
  if (g >= n8) return;
  const v4f* p = (const v4f*)(in + (size_t)g * 8);
  v4f a = p[0], b = p[1];
  v8us o;
  o[0] = f2bf(a.x); o[1] = f2bf(a.y); o[2] = f2bf(a.z); o[3] = f2bf(a.w);
  o[4] = f2bf(b.x); o[5] = f2bf(b.y); o[6] = f2bf(b.z); o[7] = f2bf(b.w);
  *(v8us*)(out + (size_t)g * 8) = o;
}

// ---------- pre-pass 2: Xe[b, e*I+i] = bf16(blend[b,e] * x[b,i]) ----------
template <int SRC_F32>
__global__ __launch_bounds__(128)
void build_xe(const void* __restrict__ src, const float* __restrict__ blend,
              unsigned short* __restrict__ xe, int I) {
  const int b = blockIdx.y;
  const int i = (blockIdx.x * 128 + threadIdx.x) * 8;
  if (i >= I) return;
  float v[8];
  if (SRC_F32) {
    const v4f* p = (const v4f*)((const float*)src + (size_t)b * I + i);
    v4f a = p[0], c = p[1];
    v[0] = a.x; v[1] = a.y; v[2] = a.z; v[3] = a.w;
    v[4] = c.x; v[5] = c.y; v[6] = c.z; v[7] = c.w;
  } else {
    v8us h = *(const v8us*)((const unsigned short*)src + (size_t)b * I + i);
#pragma unroll
    for (int j = 0; j < 8; ++j) v[j] = bf2f(h[j]);
  }
  const size_t Kp = (size_t)I * EXPERTS;
#pragma unroll
  for (int e = 0; e < EXPERTS; ++e) {
    const float s = blend[b * EXPERTS + e];
    v8us o;
#pragma unroll
    for (int j = 0; j < 8; ++j) o[j] = f2bf(v[j] * s);
    *(v8us*)(xe + (size_t)b * Kp + (size_t)e * I + i) = o;
  }
}

// ---------- TDM: 2D tile load, global -> LDS, padded rows ----------
#if HAVE_TDM
__device__ __forceinline__ void tdm_load_2d(const void* gaddr, unsigned lds_off,
                                            unsigned rows, unsigned cols,
                                            unsigned long long strideElems,
                                            unsigned tdim0, unsigned tdim1) {
  const unsigned long long ga = (unsigned long long)(uintptr_t)gaddr;
  u32x4 g0;
  g0[0] = 1u;                                   // count=1, user descriptor
  g0[1] = lds_off;                              // lds_addr (bytes)
  g0[2] = (unsigned)(ga & 0xFFFFFFFFu);         // global_addr[31:0]
  g0[3] = (unsigned)((ga >> 32) & 0x01FFFFFFu)  // global_addr[56:32]
          | 0x80000000u;                        // type=2 ("image")
  i32x8 g1;
  // data_size=2B | pad_enable | pad_interval=32 DWORDs (128B row) | pad_amount=4 DWORDs (16B)
  g1[0] = (int)((1u << 16) | (1u << 20) | (4u << 22) | (3u << 25));
  g1[1] = (int)((tdim0 & 0xFFFFu) << 16);                                  // tensor_dim0 lo
  g1[2] = (int)(((tdim0 >> 16) & 0xFFFFu) | ((tdim1 & 0xFFFFu) << 16));    // dim0 hi | dim1 lo
  g1[3] = (int)(((tdim1 >> 16) & 0xFFFFu) | ((cols & 0xFFFFu) << 16));     // dim1 hi | tile_dim0
  g1[4] = (int)(rows & 0xFFFFu);                                           // tile_dim1 (tile_dim2=0)
  g1[5] = (int)(strideElems & 0xFFFFFFFFull);                              // dim0_stride lo
  g1[6] = (int)((strideElems >> 32) & 0xFFFFull);                          // dim0_stride hi
  g1[7] = 0;
  const i32x4 z4 = {0, 0, 0, 0};
#if __has_include(<hip/amd_detail/amd_gfx1250_TDM.h>)
  const i32x8 z8 = {0, 0, 0, 0, 0, 0, 0, 0};
  __builtin_amdgcn_tensor_load_to_lds(g0, g1, z4, z4, z8, 0);
#else
  __builtin_amdgcn_tensor_load_to_lds(g0, g1, z4, z4, 0);
#endif
}
#else
struct Stg { uint4 r[4]; };
__device__ __forceinline__ Stg ld64(const unsigned short* __restrict__ g) {
  const uint4* gp = (const uint4*)g;
  Stg s; s.r[0] = gp[0]; s.r[1] = gp[1]; s.r[2] = gp[2]; s.r[3] = gp[3];
  return s;
}
__device__ __forceinline__ void st64p(unsigned short* l, int row, int half, const Stg& s) {
#pragma unroll
  for (int j = 0; j < 4; ++j)
    *(uint4*)(l + row * ROWE + (half * 4 + j) * 8) = s.r[j];
}
#endif

__device__ __forceinline__ uint4 ldsChunk(const unsigned short* l, int row, int c) {
  return *(const uint4*)(l + row * ROWE + c * 8); // 16B chunks, 144B row stride
}

// ---------- main GEMM: Y = W'(E folded into K) x Xe^T, K' = 8*I ----------
template <int ELU, int OUT_BF16>
__global__ __launch_bounds__(256)
void blended_gemm(const unsigned short* __restrict__ Xe,  // [B, 8I] bf16 (blend-scaled)
                  const unsigned short* __restrict__ Wb,  // [E,O,I] bf16
                  const float* __restrict__ Bias,         // [E,O]   f32
                  const float* __restrict__ blend,        // [B,E]   f32
                  void* __restrict__ Yv,                  // [B,O]   bf16 or f32
                  int I, int O, int kiShift) {
  extern __shared__ unsigned short smem[]; // [W0|W1|X0|X1], each TILE_E, padded rows

  const int tid  = threadIdx.x;
  const int lane = tid & 31;
  const int wid  = tid >> 5;
  const int wm   = wid >> 1;
  const int wn   = wid & 1;
  const int ml   = lane & 15;
  const int kh   = lane >> 4;

  const int oBase = blockIdx.x * BM;
  const int bBase = blockIdx.y * BN;
  const size_t Kp = (size_t)I * EXPERTS;
  const int kiMask = (1 << kiShift) - 1;
  const int nkb = (int)(Kp / BK);

  v8f acc[2][4] = {};

#if HAVE_TDM
  const unsigned ldsBase = (unsigned)__builtin_amdgcn_groupstaticsize();
  const unsigned tdW = (unsigned)((size_t)EXPERTS * O);
  if (wid == 0) { // prologue: DMA kb=0 tiles (e=0, i0=0)
    tdm_load_2d(Wb + ((size_t)oBase) * I, ldsBase + 0 * TILE_B, BM, BK,
                (unsigned long long)I, (unsigned)I, tdW);
    tdm_load_2d(Xe + (size_t)bBase * Kp, ldsBase + 2 * TILE_B, BN, BK,
                (unsigned long long)Kp, (unsigned)Kp, 4096u);
    __builtin_amdgcn_s_wait_tensorcnt(0);
  }
  __syncthreads();
#else
  const int stRow = tid >> 1;
  const int half  = tid & 1;
  const unsigned short* xRow = Xe + (size_t)(bBase + stRow) * Kp + half * 32;
  {
    Stg wS = ld64(Wb + ((size_t)oBase + stRow) * I + half * 32);
    Stg xS = ld64(xRow);
    st64p(smem + 0 * TILE_E, stRow, half, wS);
    st64p(smem + 2 * TILE_E, stRow, half, xS);
  }
  __syncthreads();
#endif

  for (int kb = 0; kb < nkb; ++kb) {
    const int p = kb & 1;
    const bool more = (kb + 1 < nkb);

#if HAVE_TDM
    if (wid == 0 && more) { // DMA next tiles into the other buffer during compute
      const int kn = kb + 1;
      const int e  = kn >> kiShift;
      const int i0 = (kn & kiMask) << 6;
      tdm_load_2d(Wb + ((size_t)e * O + oBase) * I + i0,
                  ldsBase + (unsigned)(p ^ 1) * TILE_B, BM, BK,
                  (unsigned long long)I, (unsigned)I, tdW);
      tdm_load_2d(Xe + (size_t)bBase * Kp + (size_t)kn * BK,
                  ldsBase + (unsigned)(2 + (p ^ 1)) * TILE_B, BN, BK,
                  (unsigned long long)Kp, (unsigned)Kp, 4096u);
    }
#else
    Stg wS, xS;
    if (more) {
      const int kn = kb + 1;
      const int e  = kn >> kiShift;
      const int i0 = (kn & kiMask) << 6;
      wS = ld64(Wb + ((size_t)e * O + oBase + stRow) * I + i0 + half * 32);
      xS = ld64(xRow + (size_t)kn * BK);
    }
#endif

    // compute on buffer p: two 32-deep WMMA k-steps
    const unsigned short* sWp = smem + (unsigned)p * TILE_E;
    const unsigned short* sXp = smem + (unsigned)(2 + p) * TILE_E;
#pragma unroll
    for (int ks = 0; ks < 2; ++ks) {
      v16bf afr[2], bfr[4];
#pragma unroll
      for (int mt = 0; mt < 2; ++mt) { // A: lane-half kh holds K[kh*16..+16)
        const int r = wm * 32 + mt * 16 + ml;
        afr[mt] = make_frag(ldsChunk(sWp, r, ks * 4 + kh * 2),
                            ldsChunk(sWp, r, ks * 4 + kh * 2 + 1));
      }
#pragma unroll
      for (int nt = 0; nt < 4; ++nt) { // B: lane-half kh holds K[kh*16..+16)
        const int r = wn * 64 + nt * 16 + ml;
        bfr[nt] = make_frag(ldsChunk(sXp, r, ks * 4 + kh * 2),
                            ldsChunk(sXp, r, ks * 4 + kh * 2 + 1));
      }
#pragma unroll
      for (int nt = 0; nt < 4; ++nt)
#pragma unroll
        for (int mt = 0; mt < 2; ++mt)
          acc[mt][nt] = __builtin_amdgcn_wmma_f32_16x16x32_bf16(
              false, afr[mt], false, bfr[nt], (short)0, acc[mt][nt], false, false);
    }

#if HAVE_TDM
    if (wid == 0 && more) __builtin_amdgcn_s_wait_tensorcnt(0);
#else
    if (more) {
      st64p(smem + (unsigned)(p ^ 1) * TILE_E, stRow, half, wS);
      st64p(smem + (unsigned)(2 + (p ^ 1)) * TILE_E, stRow, half, xS);
    }
#endif
    __syncthreads(); // single barrier per K step
  }

  // ---- epilogue: blended bias, activation, store ----
  float bl[4][EXPERTS];
#pragma unroll
  for (int nt = 0; nt < 4; ++nt) {
    const int b = bBase + wn * 64 + nt * 16 + ml;
#pragma unroll
    for (int e = 0; e < EXPERTS; ++e) bl[nt][e] = blend[b * EXPERTS + e];
  }
#pragma unroll
  for (int mt = 0; mt < 2; ++mt) {
#pragma unroll
    for (int r = 0; r < 8; ++r) {
      const int o = oBase + wm * 32 + mt * 16 + r + kh * 8;
      float bv[EXPERTS];
#pragma unroll
      for (int e = 0; e < EXPERTS; ++e) bv[e] = Bias[e * O + o];
#pragma unroll
      for (int nt = 0; nt < 4; ++nt) {
        float bb = 0.f;
#pragma unroll
        for (int e = 0; e < EXPERTS; ++e) bb += bl[nt][e] * bv[e];
        float y = acc[mt][nt][r] + bb;
        if (ELU) y = (y > 0.f) ? y : (__expf(y) - 1.f);
        const int b = bBase + wn * 64 + nt * 16 + ml;
        if (OUT_BF16)
          ((unsigned short*)Yv)[(size_t)b * O + o] = f2bf(y);
        else
          ((float*)Yv)[(size_t)b * O + o] = y;
      }
    }
  }
}

extern "C" void kernel_launch(void* const* d_in, const int* in_sizes, int n_in,
                              void* d_out, int out_size, void* d_ws, size_t ws_size,
                              hipStream_t stream) {
  (void)in_sizes; (void)n_in; (void)out_size; (void)ws_size;
  const float* blend = (const float*)d_in[0]; // [4096, 8]
  const float* x     = (const float*)d_in[1]; // [4096, 1024]
  const float* W0    = (const float*)d_in[2]; // [8, 2048, 1024]
  const float* B0    = (const float*)d_in[3]; // [8, 2048]
  const float* W1    = (const float*)d_in[4]; // [8, 2048, 2048]
  const float* B1    = (const float*)d_in[5]; // [8, 2048]
  const float* W2    = (const float*)d_in[6]; // [8, 512, 2048]
  const float* B2    = (const float*)d_in[7]; // [8, 512]
  float* out = (float*)d_out;                 // [4096, 512] f32

  const int Bn = 4096;
  const size_t nW0 = (size_t)EXPERTS * 2048 * 1024;
  const size_t nW1 = (size_t)EXPERTS * 2048 * 2048;
  const size_t nW2 = (size_t)EXPERTS * 512 * 2048;
  const size_t nH  = (size_t)Bn * 2048;

  // bf16 workspace layout (~272 MiB)
  unsigned short* w0b = (unsigned short*)d_ws;
  unsigned short* w1b = w0b + nW0;
  unsigned short* w2b = w1b + nW1;
  unsigned short* h0b = w2b + nW2;
  unsigned short* h1b = h0b + nH;
  unsigned short* xe  = h1b + nH; // [4096, 8*2048] bf16 max, reused per layer

  dim3 blk(256);
  cvt_f32_bf16<<<dim3((unsigned)(nW0 / 8 / 256)), blk, 0, stream>>>(W0, w0b, (int)(nW0 / 8));
  cvt_f32_bf16<<<dim3((unsigned)(nW1 / 8 / 256)), blk, 0, stream>>>(W1, w1b, (int)(nW1 / 8));
  cvt_f32_bf16<<<dim3((unsigned)(nW2 / 8 / 256)), blk, 0, stream>>>(W2, w2b, (int)(nW2 / 8));

  // layer 0: I=1024 (kiShift=4), O=2048
  build_xe<1><<<dim3(1, Bn), dim3(128), 0, stream>>>(x, blend, xe, 1024);
  blended_gemm<1, 1><<<dim3(2048 / BM, Bn / BN), blk, LDS_BYTES, stream>>>(xe, w0b, B0, blend, h0b, 1024, 2048, 4);
  // layer 1: I=2048 (kiShift=5), O=2048
  build_xe<0><<<dim3(2, Bn), dim3(128), 0, stream>>>(h0b, blend, xe, 2048);
  blended_gemm<1, 1><<<dim3(2048 / BM, Bn / BN), blk, LDS_BYTES, stream>>>(xe, w1b, B1, blend, h1b, 2048, 2048, 5);
  // layer 2: I=2048, O=512
  build_xe<0><<<dim3(2, Bn), dim3(128), 0, stream>>>(h1b, blend, xe, 2048);
  blended_gemm<0, 0><<<dim3(512 / BM, Bn / BN), blk, LDS_BYTES, stream>>>(xe, w2b, B2, blend, out, 2048, 512, 5);
}
